// pos_gnn_29497835389489
// MI455X (gfx1250) — compile-verified, matrix operand order checked by reading, not run on hardware
//
#include <hip/hip_runtime.h>

// ---------------- CDNA5 WMMA types ----------------
typedef __attribute__((ext_vector_type(16))) __bf16 v16bf;
typedef __attribute__((ext_vector_type(8)))  __bf16 v8bf;
typedef __attribute__((ext_vector_type(8)))  float  v8f;

__device__ __forceinline__ __bf16 f2bf(float f) {
    union { float f; unsigned u; } in; in.f = f;
    unsigned u = in.u;
    unsigned r = u + 0x7FFFu + ((u >> 16) & 1u);   // round-to-nearest-even
    union { unsigned short s; __bf16 b; } out; out.s = (unsigned short)(r >> 16);
    return out.b;
}

// ordered-uint encoding for float atomic max
__device__ __forceinline__ unsigned encf(float f) {
    unsigned b = __float_as_uint(f);
    return (b & 0x80000000u) ? ~b : (b | 0x80000000u);
}
__device__ __forceinline__ float decf(unsigned k) {
    unsigned b = (k & 0x80000000u) ? (k & 0x7FFFFFFFu) : ~k;
    return __uint_as_float(b);
}

// ---------------- generic fused bf16-WMMA GEMM ----------------
// C = epilogue( A' @ W + bias ), A' built per AMODE, f32 in/out, bf16 WMMA compute.
// block = 256 threads (8 wave32s). Block tile 128x128, K-step 32.
// Waves arranged 4(M) x 2(N): each wave owns a 32x64 sub-tile =
//   2 A-fragments x 4 B-fragments = 8 WMMAs per K-step, B reused 2x.
constexpr int MT = 128, NT = 128, KT = 32, LDSK = 48; // 96B row stride: 16B aligned, low-conflict

// AMODE: 0 plain A[m*lda+k]; 1 gather A[rowIdx[m]*lda+k];
//        2 pair-sum over [B=8,N=128,N=128]: A[(b,i)]+A[(b,j)]; 3 concat(A:128, A2:128, A3:256)
// EPI:   0 store out[m*Nc+n]; 1 dense-edge update (out += silu)/sqrt2;
//        2 out = tanh(aux + acc + b) (Nc=64); 3 transposed store out[b][n][i][j]
template<int AMODE, int EPI>
__global__ __launch_bounds__(256)
void gemm_k(const float* __restrict__ A, const float* __restrict__ A2,
            const float* __restrict__ A3, const int* __restrict__ rowIdx,
            const float* __restrict__ W, const float* __restrict__ bias,
            float* __restrict__ out, const float* __restrict__ aux,
            int M, int K, int Nc, int lda)
{
    __shared__ __bf16 As[MT][LDSK];
    __shared__ __bf16 Bs[NT][LDSK];
    const int tid = threadIdx.x;
    const int lane = tid & 31, w = tid >> 5;
    const int lane16 = lane & 15, half = lane >> 4;
    const int m0 = blockIdx.x * MT;
    const int n0 = blockIdx.y * NT;
    const int wm = (w & 3) * 32;        // wave row base within tile
    const int wn = (w >> 2) * 64;       // wave col base within tile

    v8f acc[2][4] = {};

    for (int kb = 0; kb < K; kb += KT) {
        // ---- stage A tile (128 x 32) as bf16 ----
        for (int idx = tid; idx < MT * KT; idx += 256) {
            int r = idx >> 5, c = idx & 31;
            int m = m0 + r, k = kb + c;
            float v;
            if (AMODE == 0) {
                v = A[(long)m * lda + k];
            } else if (AMODE == 1) {
                v = A[(long)rowIdx[m] * lda + k];
            } else if (AMODE == 2) {
                int b = m >> 14, i = (m >> 7) & 127, j = m & 127;
                v = A[(long)(b * 128 + i) * lda + k] + A[(long)(b * 128 + j) * lda + k];
            } else {
                if (k < 128)      v = A [(long)m * 128 + k];
                else if (k < 256) v = A2[(long)m * 128 + (k - 128)];
                else              v = A3[(long)m * 256 + (k - 256)];
            }
            As[r][c] = f2bf(v);
        }
        // ---- stage B tile: Bs[n][k] = W[(kb+k)*Nc + n0+n]  (128 x 32) ----
        for (int idx = tid; idx < NT * KT; idx += 256) {
            int n = idx & 127, c = idx >> 7;
            float v = (n0 + n < Nc) ? W[(long)(kb + c) * Nc + (n0 + n)] : 0.0f;
            Bs[n][c] = f2bf(v);
        }
        if (kb + KT < K)  // prefetch next W tile into caches
            __builtin_prefetch(&W[(long)(kb + KT) * Nc + n0 + (tid & 63)], 0, 1);
        __syncthreads();

        // A fragments: 16x32 bf16 each; lane holds row lane16, K groups per ISA layout
        v16bf afrag[2];
#pragma unroll
        for (int f = 0; f < 2; ++f) {
            const v8bf* alo = reinterpret_cast<const v8bf*>(&As[wm + f * 16 + lane16][half * 8]);
            const v8bf* ahi = reinterpret_cast<const v8bf*>(&As[wm + f * 16 + lane16][16 + half * 8]);
            afrag[f] = __builtin_shufflevector(*alo, *ahi,
                         0,1,2,3,4,5,6,7,8,9,10,11,12,13,14,15);
        }
#pragma unroll
        for (int t = 0; t < 4; ++t) {
            const v8bf* blo = reinterpret_cast<const v8bf*>(&Bs[wn + t * 16 + lane16][half * 8]);
            const v8bf* bhi = reinterpret_cast<const v8bf*>(&Bs[wn + t * 16 + lane16][16 + half * 8]);
            v16bf bfrag = __builtin_shufflevector(*blo, *bhi,
                            0,1,2,3,4,5,6,7,8,9,10,11,12,13,14,15);
#pragma unroll
            for (int f = 0; f < 2; ++f)
                acc[f][t] = __builtin_amdgcn_wmma_f32_16x16x32_bf16(
                              false, afrag[f], false, bfrag, (short)0, acc[f][t], false, false);
        }
        __syncthreads();
    }

    // ---- epilogue: C 16x16 layout: vgpr r -> row r+8*half, lane16 -> col ----
#pragma unroll
    for (int t = 0; t < 4; ++t) {
        int n = n0 + wn + t * 16 + lane16;
        if (n >= Nc) continue;
        float bv = bias[n];
#pragma unroll
        for (int f = 0; f < 2; ++f) {
#pragma unroll
            for (int r = 0; r < 8; ++r) {
                int m = m0 + wm + f * 16 + r + half * 8;
                float v = acc[f][t][r] + bv;
                if (EPI == 0) {
                    out[(long)m * Nc + n] = v;
                } else if (EPI == 1) {
                    long o = (long)m * 256 + n;
                    float s = v / (1.0f + __expf(-v));           // silu
                    out[o] = (out[o] + s) * 0.70710678118654752f; // /sqrt(2)
                } else if (EPI == 2) {
                    long o = (long)m * 64 + n;
                    out[o] = tanhf(aux[o] + v);
                } else {
                    int b = m >> 14, i = (m >> 7) & 127, j = m & 127;
                    out[((long)(b * 128 + n) << 14) + (i << 7) + j] = v;
                }
            }
        }
    }
}

// ---------------- small helper kernels ----------------
__global__ void concat2_k(const float* __restrict__ a, int da,
                          const float* __restrict__ b, int db,
                          float* __restrict__ o, int M) {
    int idx = blockIdx.x * 256 + threadIdx.x;
    int D = da + db;
    if (idx < M * D) {
        int m = idx / D, c = idx - m * D;
        o[idx] = (c < da) ? a[(long)m * da + c] : b[(long)m * db + (c - da)];
    }
}

__global__ void erow_k(const int* __restrict__ di, int* __restrict__ erow, int E) {
    int e = blockIdx.x * 256 + threadIdx.x;
    if (e < E) erow[e] = di[e] * 16384 + di[E + e] * 128 + di[2 * E + e];
}

__global__ void attinit_k(unsigned* __restrict__ mmax, float* __restrict__ denom,
                          float* __restrict__ outb, int nOut, int nSeg) {
    int idx = blockIdx.x * 256 + threadIdx.x;
    if (idx < nOut) outb[idx] = 0.f;
    if (idx < nSeg) { mmax[idx] = 0x007FFFFFu /* enc(-inf) */; denom[idx] = 0.f; }
}

__global__ void logits_k(const int* __restrict__ src, const int* __restrict__ dst,
                         const float* __restrict__ q, const float* __restrict__ k,
                         const float* __restrict__ ef, float* __restrict__ logits,
                         unsigned* __restrict__ mmax, int E) {
    int idx = blockIdx.x * 256 + threadIdx.x;
    if (idx >= E * 4) return;
    int e = idx >> 2, h = idx & 3;
    int s = src[e], d = dst[e];
    const float* qp = q  + (long)d * 256 + h * 64;
    const float* kp = k  + (long)s * 256 + h * 64;
    const float* ep = ef + (long)e * 256 + h * 64;
    float acc = 0.f;
#pragma unroll 8
    for (int c = 0; c < 64; ++c) acc += qp[c] * (kp[c] + ep[c]);
    float lg = acc * 0.125f;                  // 1/sqrt(HEAD_D=64)
    logits[idx] = lg;
    atomicMax(&mmax[d * 4 + h], encf(lg));
}

__global__ void expsum_k(const int* __restrict__ dst, float* __restrict__ logits,
                         const unsigned* __restrict__ mmax, float* __restrict__ denom, int E) {
    int idx = blockIdx.x * 256 + threadIdx.x;
    if (idx >= E * 4) return;
    int e = idx >> 2, h = idx & 3;
    int d = dst[e];
    float ex = __expf(logits[idx] - decf(mmax[d * 4 + h]));
    logits[idx] = ex;
    atomicAdd(&denom[d * 4 + h], ex);
}

__global__ void msg_k(const int* __restrict__ src, const int* __restrict__ dst,
                      const float* __restrict__ ex, const float* __restrict__ denom,
                      const float* __restrict__ v, const float* __restrict__ ef,
                      float* __restrict__ outb, int E) {
    int idx = blockIdx.x * 256 + threadIdx.x;
    if (idx >= E * 256) return;
    int e = idx >> 8, c = idx & 255, h = c >> 6;
    int s = src[e], d = dst[e];
    float alpha = ex[(e << 2) + h] / (denom[(d << 2) + h] + 1e-16f);
    float m = alpha * (v[(long)s * 256 + c] + ef[(long)e * 256 + c]);
    atomicAdd(&outb[(long)d * 256 + c], m);
}

// ---------------- orchestration ----------------
extern "C" void kernel_launch(void* const* d_in, const int* in_sizes, int n_in,
                              void* d_out, int out_size, void* d_ws, size_t ws_size,
                              hipStream_t stream) {
    const int Bg = 8, N = 128, E = 32768, NN = 1024, BNN = 131072;

    const float* x_degree  = (const float*)d_in[0];
    const float* x_pos     = (const float*)d_in[1];
    const int*   edge_index= (const int*)  d_in[2];
    const float* dense_ori = (const float*)d_in[3];
    const float* dense_spd = (const float*)d_in[4];
    const int*   dense_idx = (const int*)  d_in[5];
    // params flattened (sorted pytree): convs[i]: e.W,e.b,k.W,k.b,p.W,p.b,q.W,q.b,v.W,v.b
    // then edge_convs[i]: W,b ; then edge_layer: W,b
    const int* src = edge_index;
    const int* dst = edge_index + E;

    size_t off = 0;
    auto take = [&](size_t bytes) {
        void* p = (char*)d_ws + off;
        off += (bytes + 255) & ~(size_t)255;
        return p;
    };
    float*    dense_edge = (float*)take((size_t)BNN * 256 * 4);
    float*    efeat      = (float*)take((size_t)E * 256 * 4);
    float*    hc         = (float*)take((size_t)NN * 320 * 4);
    float*    qbuf       = (float*)take((size_t)NN * 256 * 4);
    float*    kbuf       = (float*)take((size_t)NN * 256 * 4);
    float*    vbuf       = (float*)take((size_t)NN * 256 * 4);
    float*    outA       = (float*)take((size_t)NN * 256 * 4);
    float*    outB       = (float*)take((size_t)NN * 256 * 4);
    float*    hpA        = (float*)take((size_t)NN * 64 * 4);
    float*    hpB        = (float*)take((size_t)NN * 64 * 4);
    float*    logitsb    = (float*)take((size_t)E * 4 * 4);
    unsigned* mmax       = (unsigned*)take((size_t)NN * 4 * 4);
    float*    denom      = (float*)take((size_t)NN * 4 * 4);
    int*      erow       = (int*)take((size_t)E * 4);
    (void)ws_size; (void)n_in; (void)in_sizes; (void)out_size; (void)Bg; (void)N;

    // dense_edge = concat(ori, spd) along channels
    concat2_k<<<(BNN * 256 + 255) / 256, 256, 0, stream>>>(dense_ori, 128, dense_spd, 128,
                                                           dense_edge, BNN);
    erow_k<<<(E + 255) / 256, 256, 0, stream>>>(dense_idx, erow, E);

    const float* hIn  = x_degree; int d1 = 128;
    const float* hpIn = x_pos;
    float* outs[2] = {outA, outB};
    float* hps[2]  = {hpA, hpB};

    for (int l = 0; l < 3; ++l) {
        int pb = 6 + l * 10;
        const float *eW = (const float*)d_in[pb+0], *eB = (const float*)d_in[pb+1];
        const float *kW = (const float*)d_in[pb+2], *kB = (const float*)d_in[pb+3];
        const float *pW = (const float*)d_in[pb+4], *pB = (const float*)d_in[pb+5];
        const float *qW = (const float*)d_in[pb+6], *qB = (const float*)d_in[pb+7];
        const float *vW = (const float*)d_in[pb+8], *vB = (const float*)d_in[pb+9];
        const float *ecW = (const float*)d_in[36 + l*2], *ecB = (const float*)d_in[37 + l*2];

        int inK = d1 + 64;                       // 192 or 320 (multiples of 32)
        float* outCur = outs[l & 1];
        float* hpOut  = hps[l & 1];

        // hc = [h | h_pos]
        concat2_k<<<(NN * inK + 255) / 256, 256, 0, stream>>>(hIn, d1, hpIn, 64, hc, NN);

        dim3 gq(NN / MT, 256 / NT);
        gemm_k<0,0><<<gq, 256, 0, stream>>>(hc, nullptr, nullptr, nullptr, qW, qB, qbuf, nullptr, NN, inK, 256, inK);
        gemm_k<0,0><<<gq, 256, 0, stream>>>(hc, nullptr, nullptr, nullptr, kW, kB, kbuf, nullptr, NN, inK, 256, inK);
        gemm_k<0,0><<<gq, 256, 0, stream>>>(hc, nullptr, nullptr, nullptr, vW, vB, vbuf, nullptr, NN, inK, 256, inK);

        // e = gather(dense_edge)[erow] @ We + be
        dim3 ge(E / MT, 256 / NT);
        gemm_k<1,0><<<ge, 256, 0, stream>>>(dense_edge, nullptr, nullptr, erow, eW, eB, efeat, nullptr, E, 256, 256, 256);

        // attention with segment softmax over dst
        attinit_k<<<(NN * 256 + 255) / 256, 256, 0, stream>>>(mmax, denom, outCur, NN * 256, NN * 4);
        logits_k<<<(E * 4 + 255) / 256, 256, 0, stream>>>(src, dst, qbuf, kbuf, efeat, logitsb, mmax, E);
        expsum_k<<<(E * 4 + 255) / 256, 256, 0, stream>>>(dst, logitsb, mmax, denom, E);
        msg_k<<<(E * 256) / 256, 256, 0, stream>>>(src, dst, logitsb, denom, vbuf, efeat, outCur, E);

        // h_pos' = tanh(h_pos + out @ Wp + bp)  (Nc=64 < NT: guarded)
        dim3 gp(NN / MT, 1);
        gemm_k<0,2><<<gp, 256, 0, stream>>>(outCur, nullptr, nullptr, nullptr, pW, pB, hpOut, hpIn, NN, 256, 64, 256);

        // dense_edge = (dense_edge + silu((h[b,i]+h[b,j]) @ Wec + bec)) / sqrt2
        dim3 gd(BNN / MT, 256 / NT);
        gemm_k<2,1><<<gd, 256, 0, stream>>>(outCur, nullptr, nullptr, nullptr, ecW, ecB, dense_edge, nullptr, BNN, 256, 256, 256);

        hIn = outCur; hpIn = hpOut; d1 = 256;
    }

    // out[b,c,i,j] = (concat(ori, spd, dense_edge) @ Wfinal + bfinal) transposed
    const float* flW = (const float*)d_in[42];
    const float* flB = (const float*)d_in[43];
    dim3 gf(BNN / MT, 1);
    gemm_k<3,3><<<gf, 256, 0, stream>>>(dense_ori, dense_spd, dense_edge, nullptr,
                                        flW, flB, (float*)d_out, nullptr, BNN, 512, 128, 0);
}
